// MCGRU_45921790329248
// MI455X (gfx1250) — matrix operand order repaired and checked by reading, not compiled, not run on hardware
//
#include <hip/hip_runtime.h>

#define BS   256
#define TT   128
#define LAB  64
#define DEMO 32
#define HID  128
#define FEAT 32
#define G3   96   // 3*FEAT

typedef __attribute__((ext_vector_type(16))) _Float16 v16h;
typedef __attribute__((ext_vector_type(8)))  _Float16 v8h;
typedef __attribute__((ext_vector_type(8)))  float    v8f;

__device__ __forceinline__ float fast_sigmoid(float x) {
    return __builtin_amdgcn_rcpf(1.0f + __expf(-x));
}
__device__ __forceinline__ float fast_tanh(float x) {
    return fmaf(2.0f, __builtin_amdgcn_rcpf(1.0f + __expf(-2.0f * x)), -1.0f);
}

// ---------------------------------------------------------------------------
// Kernel 1: xp[l][b][t] = lab_b[l] + sum_k x[b,t,k]*lab_W[l,k]
// channel-major layout so the GRU kernel's per-block slice is contiguous.
// ---------------------------------------------------------------------------
__global__ __launch_bounds__(256) void lab_proj_kernel(
    const float* __restrict__ x, const float* __restrict__ lab_W,
    const float* __restrict__ lab_b, float* __restrict__ xp_t)
{
    int tid = blockIdx.x * blockDim.x + threadIdx.x;   // linear (l, b, t)
    int l   = tid / (BS * TT);
    int rem = tid % (BS * TT);
    int b   = rem / TT;
    int t   = rem % TT;
    const float* xr = x + (b * TT + t) * LAB;
    const float* wr = lab_W + l * LAB;
    float acc = lab_b[l];
#pragma unroll 8
    for (int k = 0; k < LAB; ++k) acc = fmaf(xr[k], wr[k], acc);
    xp_t[tid] = acc;
}

// ---------------------------------------------------------------------------
// Kernel 2: last[b] = clip(sum_t mask[b,t] - 1, 0, T-1)
// ---------------------------------------------------------------------------
__global__ void last_kernel(const int* __restrict__ mask, int* __restrict__ last)
{
    int b = threadIdx.x;
    int s = 0;
#pragma unroll 8
    for (int t = 0; t < TT; ++t) s += mask[b * TT + t];
    s -= 1;
    if (s < 0) s = 0;
    if (s > TT - 1) s = TT - 1;
    last[b] = s;
}

// ---------------------------------------------------------------------------
// Kernel 3: multi-channel GRU recurrence via v_wmma_f32_16x16x32_f16.
// Block = (channel l, 64 batch rows); 4 waves x 32 lanes; wave owns 16 rows.
// Per step per wave: 6 WMMAs (96 gate cols), gate VALU, cndmask snapshot.
// ---------------------------------------------------------------------------
__global__ __launch_bounds__(128) void mcgru_kernel(
    const float* __restrict__ xp_t,   // [LAB][BS][TT]
    const float* __restrict__ Wih,    // [LAB][96]
    const float* __restrict__ bih,    // [LAB][96]
    const float* __restrict__ Whh,    // [LAB][96][32]
    const float* __restrict__ bhh,    // [LAB][96]
    const int*   __restrict__ last,   // [BS]
    float* __restrict__ hfinal)       // [BS][LAB][FEAT]
{
    __shared__ __align__(64) _Float16 sWhh[G3][FEAT];      // 6 KB, f16 weights
    __shared__ __align__(64) _Float16 sH[4][16][FEAT];     // 4 KB, per-wave h staging
    __shared__ __align__(64) float    sXp[64][TT];         // 32 KB, xp tile
    __shared__ float sWih_[G3], sBih_[G3], sBhh_[G3];

    const int l    = blockIdx.x;
    const int b0   = blockIdx.y * 64;
    const int tid  = threadIdx.x;
    const int lane = tid & 31;
    const int wv   = tid >> 5;
    const int nIdx = lane & 15;     // N index / row index
    const int kh   = lane >> 4;     // K-half selector

    // --- stage weights (f32 -> f16) and biases into LDS ---
    for (int i = tid; i < G3 * FEAT; i += 128)
        (&sWhh[0][0])[i] = (_Float16)Whh[l * G3 * FEAT + i];
    if (tid < G3) {
        sWih_[tid] = Wih[l * G3 + tid];
        sBih_[tid] = bih[l * G3 + tid];
        sBhh_[tid] = bhh[l * G3 + tid];
    }
    // --- stage xp tile (contiguous, coalesced b128 loads) ---
    {
        const float4* s4 = (const float4*)(xp_t + (size_t)(l * BS + b0) * TT);
        float4* d4 = (float4*)&sXp[0][0];
        for (int i = tid; i < 64 * TT / 4; i += 128) d4[i] = s4[i];
    }
    __syncthreads();

    // --- B fragments: tile jt covers gate cols [16jt,16jt+16), K=0..31 ---
    v16h bfrag[6];
#pragma unroll
    for (int jt = 0; jt < 6; ++jt) {
        v8h blo = *(const v8h*)&sWhh[jt * 16 + nIdx][kh * 8];
        v8h bhi = *(const v8h*)&sWhh[jt * 16 + nIdx][16 + kh * 8];
        bfrag[jt] = __builtin_shufflevector(blo, bhi, 0,1,2,3,4,5,6,7,8,9,10,11,12,13,14,15);
    }

    // --- per-lane gate constants (bhh folds into bias for r,z; C-init for n) ---
    float wr_[2], wz_[2], wn_[2], cr_[2], cz_[2], cnx_[2];
    v8f cinitN[2];
#pragma unroll
    for (int hf = 0; hf < 2; ++hf) {
        int f = nIdx + 16 * hf;
        wr_[hf] = sWih_[f];      cr_[hf]  = sBih_[f] + sBhh_[f];
        wz_[hf] = sWih_[32 + f]; cz_[hf]  = sBih_[32 + f] + sBhh_[32 + f];
        wn_[hf] = sWih_[64 + f]; cnx_[hf] = sBih_[64 + f];
        float bn = sBhh_[64 + f];
#pragma unroll
        for (int e = 0; e < 8; ++e) cinitN[hf][e] = bn;
    }

    int lastv[8];
#pragma unroll
    for (int j = 0; j < 8; ++j) lastv[j] = last[b0 + wv * 16 + j + 8 * kh];

    v8f hA = {}, hB = {}, snA = {}, snB = {};
    const v8f zero = {};

#pragma unroll 1
    for (int t = 0; t < TT; ++t) {
        // h (f32 C-layout) -> f16 staging -> 16-bit A-layout fragment.
        // Same-wave DS ops are in-order; region is wave-private (no barrier).
#pragma unroll
        for (int j = 0; j < 8; ++j) {
            int m = j + 8 * kh;
            sH[wv][m][nIdx]      = (_Float16)hA[j];
            sH[wv][m][nIdx + 16] = (_Float16)hB[j];
        }
        v8h alo = *(const v8h*)&sH[wv][nIdx][kh * 8];        // K 0..7 / 8..15
        v8h ahi = *(const v8h*)&sH[wv][nIdx][16 + kh * 8];   // K 16..23 / 24..31
        v16h a = __builtin_shufflevector(alo, ahi,
                 0,1,2,3,4,5,6,7,8,9,10,11,12,13,14,15);

        float xv[8];
#pragma unroll
        for (int j = 0; j < 8; ++j)
            xv[j] = sXp[wv * 16 + j + 8 * kh][t];            // same-addr broadcast

#pragma unroll
        for (int hf = 0; hf < 2; ++hf) {
            v8f cR = __builtin_amdgcn_wmma_f32_16x16x32_f16(
                false, a, false, bfrag[0 + hf], (short)0, zero, false, false);
            v8f cZ = __builtin_amdgcn_wmma_f32_16x16x32_f16(
                false, a, false, bfrag[2 + hf], (short)0, zero, false, false);
            v8f cN = __builtin_amdgcn_wmma_f32_16x16x32_f16(
                false, a, false, bfrag[4 + hf], (short)0, cinitN[hf], false, false);
            v8f& h  = hf ? hB : hA;
            v8f& sn = hf ? snB : snA;
#pragma unroll
            for (int j = 0; j < 8; ++j) {
                float r  = fast_sigmoid(fmaf(xv[j], wr_[hf], cr_[hf]) + cR[j]);
                float z  = fast_sigmoid(fmaf(xv[j], wz_[hf], cz_[hf]) + cZ[j]);
                float n  = fast_tanh(fmaf(xv[j], wn_[hf], cnx_[hf]) + r * cN[j]);
                float hn = fmaf(z, h[j] - n, n);             // (1-z)*n + z*h
                h[j]  = hn;
                sn[j] = (t == lastv[j]) ? hn : sn[j];        // snapshot at last[b]
            }
        }
    }

    // --- write snapshots: hfinal[b][l][f] ---
#pragma unroll
    for (int j = 0; j < 8; ++j) {
        int b = b0 + wv * 16 + j + 8 * kh;
        float* dst = hfinal + ((size_t)b * LAB + l) * FEAT;
        dst[nIdx]      = snA[j];
        dst[nIdx + 16] = snB[j];
    }
}

// ---------------------------------------------------------------------------
// Kernel 4: out[b,:] = [demo(static), hfinal(b)] @ out_W^T + out_b
// Only at the selected timestep -> 256 x (2176->128). One block per b.
// ---------------------------------------------------------------------------
__global__ __launch_bounds__(128) void out_proj_kernel(
    const float* __restrict__ stat,  const float* __restrict__ demo_W,
    const float* __restrict__ demo_b, const float* __restrict__ hfinal,
    const float* __restrict__ out_W, const float* __restrict__ out_b,
    float* __restrict__ out)
{
    __shared__ float sStat[DEMO];
    __shared__ float sDemo[HID];
    __shared__ float sHf[LAB * FEAT];
    const int b = blockIdx.x;
    const int tid = threadIdx.x;

    if (tid < DEMO) sStat[tid] = stat[b * DEMO + tid];
    __syncthreads();
    {
        float acc = demo_b[tid];
        const float* w = demo_W + tid * DEMO;
#pragma unroll
        for (int k = 0; k < DEMO; ++k) acc = fmaf(sStat[k], w[k], acc);
        sDemo[tid] = acc;
    }
    for (int i = tid; i < LAB * FEAT; i += HID)
        sHf[i] = hfinal[(size_t)b * LAB * FEAT + i];
    __syncthreads();

    float acc = out_b[tid];
    const float* w = out_W + (size_t)tid * (HID + LAB * FEAT);
#pragma unroll 4
    for (int k = 0; k < HID; ++k) acc = fmaf(sDemo[k], w[k], acc);
    const float* w2 = w + HID;
#pragma unroll 8
    for (int k = 0; k < LAB * FEAT; ++k) acc = fmaf(sHf[k], w2[k], acc);
    out[b * HID + tid] = acc;
}

// ---------------------------------------------------------------------------
extern "C" void kernel_launch(void* const* d_in, const int* in_sizes, int n_in,
                              void* d_out, int out_size, void* d_ws, size_t ws_size,
                              hipStream_t stream)
{
    (void)in_sizes; (void)n_in; (void)out_size; (void)ws_size;
    const float* x      = (const float*)d_in[0];
    const float* stat   = (const float*)d_in[1];
    const int*   mask   = (const int*)d_in[2];
    const float* demo_W = (const float*)d_in[3];
    const float* demo_b = (const float*)d_in[4];
    const float* lab_W  = (const float*)d_in[5];
    const float* lab_b  = (const float*)d_in[6];
    const float* Wih    = (const float*)d_in[7];
    const float* bih    = (const float*)d_in[8];
    const float* Whh    = (const float*)d_in[9];
    const float* bhh    = (const float*)d_in[10];
    const float* out_W  = (const float*)d_in[11];
    const float* out_b  = (const float*)d_in[12];

    char* ws = (char*)d_ws;
    const size_t XP_BYTES = (size_t)LAB * BS * TT * sizeof(float);   // 8 MB
    float* xp_t   = (float*)ws;
    int*   lastp  = (int*)(ws + XP_BYTES);
    float* hfinal = (float*)(ws + XP_BYTES + 4096);

    lab_proj_kernel<<<(LAB * BS * TT) / 256, 256, 0, stream>>>(x, lab_W, lab_b, xp_t);
    last_kernel<<<1, BS, 0, stream>>>(mask, lastp);
    mcgru_kernel<<<dim3(LAB, BS / 64), 128, 0, stream>>>(
        xp_t, Wih, bih, Whh, bhh, lastp, hfinal);
    out_proj_kernel<<<BS, HID, 0, stream>>>(
        stat, demo_W, demo_b, hfinal, out_W, out_b, (float*)d_out);
}
